// HyperAggregationBatched_68143951118650
// MI455X (gfx1250) — compile-verified, matrix operand order checked by reading, not run on hardware
//
#include <hip/hip_runtime.h>
#include <hip/hip_bf16.h>
#include <cmath>

// CDNA5 WMMA fragment types (wave32): 16x32 f16 A/B = 16 halves/lane,
// 16x16 f32 C/D = 8 floats/lane.
typedef __attribute__((ext_vector_type(16))) _Float16 v16h;
typedef __attribute__((ext_vector_type(8)))  float    v8f;
// Async-LDS builtin wants pointers to 16-byte int vectors.
typedef int v4i_vs __attribute__((vector_size(4 * sizeof(int))));

#define HIDDEN 128
#define MIX    8
#define MTILE  32      // edges per wave (2 x 16-row WMMA M-tiles)
#define WPB    2       // waves per block in the WMMA kernel

#define AS1 __attribute__((address_space(1)))
#define AS3 __attribute__((address_space(3)))

#if __has_builtin(__builtin_amdgcn_global_load_async_to_lds_b128)
#define HAVE_ASYNC 1
#else
#define HAVE_ASYNC 0
#endif

__device__ __forceinline__ void async_copy16(const void* g, void* lds) {
#if HAVE_ASYNC
    // gfx1250 GLOBAL_LOAD_ASYNC_TO_LDS_B128: ASYNCcnt-tracked, no VGPR round trip.
    // Low 32 bits of a generic shared pointer are the LDS byte address.
    __builtin_amdgcn_global_load_async_to_lds_b128(
        (AS1 v4i_vs*)(uintptr_t)g,
        (AS3 v4i_vs*)(uint32_t)(uintptr_t)lds, 0, 0);
#else
    *(uint4*)lds = *(const uint4*)g;
#endif
}

__device__ __forceinline__ void async_wait0() {
#if HAVE_ASYNC
#if __has_builtin(__builtin_amdgcn_s_wait_asynccnt)
    __builtin_amdgcn_s_wait_asynccnt(0);
#else
    asm volatile("s_wait_asynccnt 0x0" ::: "memory");
#endif
#endif
}

__device__ __forceinline__ float gelu_exact(float x) {
    return 0.5f * x * (1.0f + erff(x * 0.7071067811865475f));
}

// Load one 16-half WMMA fragment (A or B) from an LDS row already offset by
// kk + hi*8: elements 0..7 = K+0..7, elements 8..15 = K+16..23 (ISA striping).
__device__ __forceinline__ v16h load_frag(const _Float16* rowbase) {
    v16h f;
    #pragma unroll
    for (int i = 0; i < 8; ++i) f[i]     = rowbase[i];
    #pragma unroll
    for (int i = 0; i < 8; ++i) f[8 + i] = rowbase[16 + i];
    return f;
}

// ---------------------------------------------------------------- kernel 0
__global__ void k_zero(float* __restrict__ p, int n) {
    int i = blockIdx.x * blockDim.x + threadIdx.x;
    int stride = gridDim.x * blockDim.x;
    for (; i < n; i += stride) p[i] = 0.0f;
}

// ---------------------------------------------------------------- kernel 0b
// One-shot f32 -> f16 conversion of X, W_A, W_B so all later staging is pure
// byte movement (async LDS copies) and gather traffic is halved.
__global__ void k_cvt(const float* __restrict__ X,
                      const float* __restrict__ W_A,
                      const float* __restrict__ W_B,
                      _Float16* __restrict__ Xh,
                      _Float16* __restrict__ WAB,   // [128*128 + 8*128]
                      int nX) {
    int stride = gridDim.x * blockDim.x;
    for (int i = blockIdx.x * blockDim.x + threadIdx.x; i < nX; i += stride)
        Xh[i] = (_Float16)X[i];
    for (int i = blockIdx.x * blockDim.x + threadIdx.x; i < HIDDEN * HIDDEN; i += stride)
        WAB[i] = (_Float16)W_A[i];
    for (int i = blockIdx.x * blockDim.x + threadIdx.x; i < MIX * HIDDEN; i += stride)
        WAB[HIDDEN * HIDDEN + i] = (_Float16)W_B[i];
}

// ---------------------------------------------------------------- kernel 1
// W_tar[e][m] = (gelu(X[src[e]] @ W_A^T) @ W_B^T)[m]
// One wave handles 32 edges (two 16-row M-tiles sharing each B fragment).
__global__ __launch_bounds__(WPB * 32)
void k_wtar(const _Float16* __restrict__ Xh,
            const _Float16* __restrict__ WAB,
            const int*      __restrict__ edge_index,   // [2*E], src = [0..E)
            float*          __restrict__ wtar,         // [E * MIX]
            int n_edges) {
    // 34 KB weights + WPB * 8 KB staging = 50 KB (< 64 KB CU-mode static cap)
    __shared__ __align__(16) _Float16 sW[HIDDEN * HIDDEN + MIX * HIDDEN];
    __shared__ __align__(16) _Float16 sT[WPB][MTILE * HIDDEN];

    const int tid  = threadIdx.x;
    const int lane = tid & 31;
    const int wave = tid >> 5;

    // ---- stage W_A^T/W_B^T (f16) into LDS: async 16B copies
    {
        const int n16 = (HIDDEN * HIDDEN + MIX * HIDDEN) * 2 / 16;  // 2176
        for (int i = tid; i < n16; i += blockDim.x)
            async_copy16((const char*)WAB + i * 16, (char*)sW + i * 16);
        async_wait0();
    }
    __syncthreads();

    const int  tile = blockIdx.x * WPB + wave;
    const long e0   = (long)tile * MTILE;
    if (e0 >= n_edges) return;

    _Float16* sX = sT[wave];
    const _Float16* sWB = sW + HIDDEN * HIDDEN;

    // ---- gather 32 rows of Xh into LDS (16 lanes x 16B per row, async)
    for (int r = 0; r < MTILE; ++r) {
        long ee = e0 + r;
        int  s  = edge_index[ee < n_edges ? ee : 0];   // src node
        if (lane < 16)
            async_copy16(Xh + (long)s * HIDDEN + lane * 8,
                         sX + r * HIDDEN + lane * 8);
    }
    async_wait0();
    // wave-local LDS RAW after wait: DS ops from one wave are serviced in order.

    const int col = lane & 15;   // A-row index / B,C,D column index
    const int hi  = lane >> 4;   // half-wave select for the K striping

    // ---- GEMM1: H(32x128) = Xs(32x128) @ W_A^T; 2 M-tiles x 8 N-tiles x 4 K
    v8f acc[2][8];
    #pragma unroll
    for (int mt = 0; mt < 2; ++mt)
        #pragma unroll
        for (int nt = 0; nt < 8; ++nt) acc[mt][nt] = (v8f){0,0,0,0,0,0,0,0};

    #pragma unroll
    for (int kk = 0; kk < HIDDEN; kk += 32) {
        const v16h a0 = load_frag(sX + col * HIDDEN + kk + hi * 8);
        const v16h a1 = load_frag(sX + (16 + col) * HIDDEN + kk + hi * 8);
        #pragma unroll
        for (int nt = 0; nt < 8; ++nt) {
            const v16h b = load_frag(sW + (nt * 16 + col) * HIDDEN + kk + hi * 8);
            acc[0][nt] = __builtin_amdgcn_wmma_f32_16x16x32_f16(
                false, a0, false, b, (short)0, acc[0][nt], false, false);
            acc[1][nt] = __builtin_amdgcn_wmma_f32_16x16x32_f16(
                false, a1, false, b, (short)0, acc[1][nt], false, false);
        }
    }

    // ---- gelu + transpose H back into LDS (D layout -> row-major f16)
    #pragma unroll
    for (int mt = 0; mt < 2; ++mt)
        #pragma unroll
        for (int nt = 0; nt < 8; ++nt) {
            int n = nt * 16 + col;
            #pragma unroll
            for (int r = 0; r < 8; ++r) {
                int mm = mt * 16 + r + hi * 8;
                sX[mm * HIDDEN + n] = (_Float16)gelu_exact(acc[mt][nt][r]);
            }
        }

    // ---- GEMM2: Wt(32x8) = H(32x128) @ W_B^T (N-tile zero-padded to 16)
    v8f aw[2] = {(v8f){0,0,0,0,0,0,0,0}, (v8f){0,0,0,0,0,0,0,0}};
    #pragma unroll
    for (int kk = 0; kk < HIDDEN; kk += 32) {
        const v16h a0 = load_frag(sX + col * HIDDEN + kk + hi * 8);
        const v16h a1 = load_frag(sX + (16 + col) * HIDDEN + kk + hi * 8);
        v16h b;
        if (col < MIX) {
            b = load_frag(sWB + col * HIDDEN + kk + hi * 8);
        } else {
            #pragma unroll
            for (int i = 0; i < 16; ++i) b[i] = (_Float16)0.0f;
        }
        aw[0] = __builtin_amdgcn_wmma_f32_16x16x32_f16(
            false, a0, false, b, (short)0, aw[0], false, false);
        aw[1] = __builtin_amdgcn_wmma_f32_16x16x32_f16(
            false, a1, false, b, (short)0, aw[1], false, false);
    }

    // ---- store W_tar (f32): lane column = mix index, rows = edges
    if (col < MIX) {
        #pragma unroll
        for (int mt = 0; mt < 2; ++mt)
            #pragma unroll
            for (int r = 0; r < 8; ++r) {
                long e = e0 + mt * 16 + r + hi * 8;
                if (e < n_edges) wtar[e * MIX + col] = aw[mt][r];
            }
    }
}

// ---------------------------------------------------------------- kernel 2
// agg[dst[e]][h][m] += X[src[e]][h] * W_tar[e][m]   (fp32 atomics, L2-resident)
__global__ __launch_bounds__(256)
void k_scatter(const float* __restrict__ X,
               const int*   __restrict__ edge_index,
               const float* __restrict__ wtar,
               float*       __restrict__ agg,
               int n_edges) {
    int e = blockIdx.x;
    int s = edge_index[e];
    int d = edge_index[n_edges + e];
    float*       bin  = agg + (long)d * (HIDDEN * MIX);
    const float* xrow = X + (long)s * HIDDEN;
    const float* wt   = wtar + (long)e * MIX;
    #pragma unroll
    for (int i = 0; i < (HIDDEN * MIX) / 256; ++i) {   // 4 iterations
        int idx = threadIdx.x + i * 256;
        float v = xrow[idx >> 3] * wt[idx & 7];
        atomicAdd(&bin[idx], v);
    }
}

// ---------------------------------------------------------------- kernel 3
__global__ void k_gelu(float* __restrict__ p, int n) {
    int i = blockIdx.x * blockDim.x + threadIdx.x;
    int stride = gridDim.x * blockDim.x;
    for (; i < n; i += stride) p[i] = gelu_exact(p[i]);
}

// ---------------------------------------------------------------- kernel 4
// out_edge[e][h] = sum_m agg[dst[e]][h][m] * W_tar[e][m]
__global__ __launch_bounds__(HIDDEN)
void k_gather(const float* __restrict__ agg,
              const int*   __restrict__ edge_index,
              const float* __restrict__ wtar,
              float*       __restrict__ out,
              int n_edges) {
    int e = blockIdx.x;
    int h = threadIdx.x;
    int d = edge_index[n_edges + e];
    const float* arow = agg + (long)d * (HIDDEN * MIX) + h * MIX;
    const float* wt   = wtar + (long)e * MIX;
    float sum = 0.0f;
    #pragma unroll
    for (int m = 0; m < MIX; ++m) sum += arow[m] * wt[m];
    out[(long)e * HIDDEN + h] = sum;
}

// ---------------------------------------------------------------- kernel 5
// Tuple tail: src / dst written as floats after out_edge.
__global__ void k_idx(const int* __restrict__ edge_index,
                      float* __restrict__ out_tail, int n_edges) {
    int i = blockIdx.x * blockDim.x + threadIdx.x;
    int stride = gridDim.x * blockDim.x;
    for (; i < n_edges; i += stride) {
        out_tail[i]           = (float)edge_index[i];            // src
        out_tail[n_edges + i] = (float)edge_index[n_edges + i];  // dst
    }
}

// ----------------------------------------------------------------
extern "C" void kernel_launch(void* const* d_in, const int* in_sizes, int n_in,
                              void* d_out, int out_size, void* d_ws, size_t ws_size,
                              hipStream_t stream) {
    const float* X   = (const float*)d_in[0];
    const float* W_A = (const float*)d_in[1];
    const float* W_B = (const float*)d_in[2];
    const int*   ei  = (const int*)d_in[3];

    const int n_nodes = in_sizes[0] / HIDDEN;
    const int n_edges = in_sizes[3] / 2;

    // workspace layout (all 16B-aligned):
    float*    agg  = (float*)d_ws;                                   // N*128*8 f32
    float*    wtar = agg + (size_t)n_nodes * HIDDEN * MIX;           // E*8 f32
    _Float16* Xh   = (_Float16*)(wtar + (size_t)n_edges * MIX);      // N*128 f16
    _Float16* WAB  = Xh + (size_t)n_nodes * HIDDEN;                  // 17408 f16

    float* out_edge = (float*)d_out;
    float* out_tail = out_edge + (size_t)n_edges * HIDDEN;

    const int agg_n = n_nodes * HIDDEN * MIX;

    // 0: zero accumulator bins (every call; replays don't re-poison)
    k_zero<<<1024, 256, 0, stream>>>(agg, agg_n);

    // 0b: f32 -> f16 staging copies of X / W_A / W_B
    k_cvt<<<1024, 256, 0, stream>>>(X, W_A, W_B, Xh, WAB, n_nodes * HIDDEN);

    // 1: hyper-network weights per edge (WMMA, async LDS staging)
    {
        int tiles  = (n_edges + MTILE - 1) / MTILE;
        int blocks = (tiles + WPB - 1) / WPB;
        k_wtar<<<blocks, WPB * 32, 0, stream>>>(Xh, WAB, ei, wtar, n_edges);
    }

    // 2: edge outer-product scatter-add
    k_scatter<<<n_edges, 256, 0, stream>>>(X, ei, wtar, agg, n_edges);

    // 3: gelu on bins
    k_gelu<<<1024, 256, 0, stream>>>(agg, agg_n);

    // 4: gather + mix contraction
    k_gather<<<n_edges, HIDDEN, 0, stream>>>(agg, ei, wtar, out_edge, n_edges);

    // 5: index outputs
    k_idx<<<512, 256, 0, stream>>>(ei, out_tail, n_edges);
}
